// RnnGcn_19808389169899
// MI455X (gfx1250) — matrix operand order repaired and checked by reading, not compiled, not run on hardware
//
#include <hip/hip_runtime.h>
#include <hip/hip_bf16.h>

typedef __bf16 bf16_t;
typedef __bf16 v16bf __attribute__((ext_vector_type(16)));
typedef __bf16 v8bf  __attribute__((ext_vector_type(8)));
typedef float  v8f   __attribute__((ext_vector_type(8)));

#define V_VOCAB 32000
#define D_IN    300
#define D_PAD   320
#define NNODE   4096
#define TSEQ    64
#define NB      64
#define NPG     64
#define NEDGE   65536
#define RH      256
#define H3      768
#define GHID    256

// register blocking: each wave computes (MT*16) x (NT*16) of C
#define MT 4
#define NT 2

// ---------------- packing / utility kernels ----------------

__global__ void pack_rows_bf16(const float* __restrict__ src, bf16_t* __restrict__ dst,
                               int rows, int cols, int ldd) {
  int idx = blockIdx.x * blockDim.x + threadIdx.x;
  if (idx >= rows * ldd) return;
  int r = idx / ldd, c = idx % ldd;
  float v = (c < cols) ? src[(size_t)r * cols + c] : 0.0f;
  dst[idx] = (bf16_t)v;
}

__global__ void pack_transpose_bf16(const float* __restrict__ src, bf16_t* __restrict__ dst,
                                    int K, int Ncol) {
  // src is [K, Ncol] row-major; dst is [Ncol, K] row-major
  int idx = blockIdx.x * blockDim.x + threadIdx.x;
  if (idx >= K * Ncol) return;
  int n = idx / K, k = idx % K;
  dst[idx] = (bf16_t)src[(size_t)k * Ncol + n];
}

__global__ void fill_f32(float* p, int n) {
  int i = blockIdx.x * blockDim.x + threadIdx.x;
  if (i < n) p[i] = 0.0f;
}

__global__ void fill_bf16(bf16_t* p, int n) {
  int i = blockIdx.x * blockDim.x + threadIdx.x;
  if (i < n) p[i] = (bf16_t)0.0f;
}

__global__ void cast_f32_bf16(const float* __restrict__ s, bf16_t* __restrict__ d, int n) {
  int i = blockIdx.x * blockDim.x + threadIdx.x;
  if (i < n) d[i] = (bf16_t)s[i];
}

// ---------------- WMMA bf16 GEMM: C[M,Nc] = act(A[M,K] @ B^T + bias) ----------------
// A: [M,K] bf16 (row gathered through `gather` if non-null: arow = gather[row*gstride+goff])
// B: [Nc,K] bf16 row-major (operand pre-transposed so K is contiguous)
// Each wave computes a 64x32 block: 4x2 grid of 16x16 WMMA tiles.
// K loop unrolled x2 with ping-pong fragment buffers (software pipelining): loads for
// step s+1 are in flight while step s's 8 WMMAs execute.

__device__ __forceinline__ void load_frags(const bf16_t* const* ap, const bf16_t* const* bp,
                                           int k0, int half, v16bf* a, v16bf* b) {
#pragma unroll
  for (int mi = 0; mi < MT; ++mi) {
    a[mi] = *(const v16bf*)(ap[mi] + k0 + half * 16);       // K = k0 + half*16 + 0..15
  }
#pragma unroll
  for (int ni = 0; ni < NT; ++ni)
    b[ni] = *(const v16bf*)(bp[ni] + k0);                    // K = k0 + half*16 + 0..15
}

__device__ __forceinline__ void do_wmma(const v16bf* a, const v16bf* b, v8f* acc) {
#pragma unroll
  for (int mi = 0; mi < MT; ++mi)
#pragma unroll
    for (int ni = 0; ni < NT; ++ni)
      acc[mi * NT + ni] = __builtin_amdgcn_wmma_f32_16x16x32_bf16(
          false, a[mi], false, b[ni], (short)0, acc[mi * NT + ni], false, false);
}

__global__ void wmma_gemm_bf16(const bf16_t* __restrict__ A, int lda,
                               const int* __restrict__ gather, int gstride, int goff,
                               const bf16_t* __restrict__ B, int ldb,
                               const float* __restrict__ bias,
                               float* __restrict__ C, int ldc,
                               int M, int Ncols, int K, int relu) {
  int wid  = blockIdx.x * 8 + (threadIdx.x >> 5);
  int lane = threadIdx.x & 31;
  int bn = Ncols >> (4 + 1);        // blocks of NT=2 tiles in N
  int bm = M >> (4 + 2);            // blocks of MT=4 tiles in M
  if (wid >= bm * bn) return;
  int tm = (wid / bn) * MT;         // base 16-tile index in M
  int tn = (wid % bn) * NT;         // base 16-tile index in N

  int l15  = lane & 15;
  int half = lane >> 4;             // 0: lanes 0-15, 1: lanes 16-31

  // A fragment source rows (per ISA 16-bit A 16x32 layout)
  const bf16_t* ap[MT];
#pragma unroll
  for (int mi = 0; mi < MT; ++mi) {
    int row  = (tm + mi) * 16 + l15;
    int arow = gather ? gather[(size_t)row * gstride + goff] : row;
    ap[mi] = A + (size_t)arow * lda;
  }
  // B fragments: column = l15, K-half by lane group (contiguous k = 16*half + s, same order as A)
  const bf16_t* bp[NT];
#pragma unroll
  for (int ni = 0; ni < NT; ++ni) {
    int ncol = (tn + ni) * 16 + l15;
    bp[ni] = B + (size_t)ncol * ldb + half * 16;
  }

  v8f acc[MT * NT];
#pragma unroll
  for (int i = 0; i < MT * NT; ++i) acc[i] = (v8f){};

  // Software-pipelined K loop. steps = K/32 is even for all uses (K in {256,320,512}).
  int steps = K >> 5;
  v16bf a0[MT], b0[NT], a1[MT], b1[NT];
  load_frags(ap, bp, 0, half, a0, b0);
  for (int s = 0; s < steps; s += 2) {
    load_frags(ap, bp, (s + 1) << 5, half, a1, b1);
    do_wmma(a0, b0, acc);
    if (s + 2 < steps) load_frags(ap, bp, (s + 2) << 5, half, a0, b0);
    do_wmma(a1, b1, acc);
  }

#pragma unroll
  for (int ni = 0; ni < NT; ++ni) {
    int ncol = (tn + ni) * 16 + l15;
    float bv = bias ? bias[ncol] : 0.0f;
#pragma unroll
    for (int mi = 0; mi < MT; ++mi) {
#pragma unroll
      for (int r = 0; r < 8; ++r) {
        int rm = (tm + mi) * 16 + r + half * 8;   // per ISA 32-bit C/D 16x16 layout
        float v = acc[mi * NT + ni][r] + bv;
        if (relu) v = fmaxf(v, 0.0f);
        C[(size_t)rm * ldc + ncol] = v;
      }
    }
  }
}

// ---------------- GRU gate kernels ----------------

__device__ __forceinline__ float sigm(float x) { return 1.0f / (1.0f + __expf(-x)); }

__global__ void gru_gate_step(const float* __restrict__ xp, const float* __restrict__ gh,
                              float* __restrict__ h_f32, bf16_t* __restrict__ h_b16) {
  int idx = blockIdx.x * blockDim.x + threadIdx.x;
  if (idx >= NNODE * RH) return;
  int n = idx >> 8, j = idx & 255;
  const float* x = xp + (size_t)n * H3;
  const float* g = gh + (size_t)n * H3;
  float r  = sigm(x[j] + g[j]);
  float z  = sigm(x[j + RH] + g[j + RH]);
  float nn = tanhf(x[j + 2 * RH] + r * g[j + 2 * RH]);
  float hp = h_f32[idx];
  float h  = (1.0f - z) * nn + z * hp;
  h_f32[idx] = h;
  h_b16[idx] = (bf16_t)h;
}

__global__ void gru_gate_bwd(const float* __restrict__ xp, const float* __restrict__ bhh,
                             float* __restrict__ hcat) {
  int idx = blockIdx.x * blockDim.x + threadIdx.x;
  if (idx >= NNODE * RH) return;
  int n = idx >> 8, j = idx & 255;
  const float* x = xp + (size_t)n * H3;
  float r  = sigm(x[j] + bhh[j]);
  float z  = sigm(x[j + RH] + bhh[j + RH]);
  float nn = tanhf(x[j + 2 * RH] + r * bhh[j + 2 * RH]);
  hcat[(size_t)n * 512 + 256 + j] = (1.0f - z) * nn;   // h_prev == 0
}

__global__ void copy_fwd_to_cat(const float* __restrict__ h, float* __restrict__ hcat) {
  int idx = blockIdx.x * blockDim.x + threadIdx.x;
  if (idx >= NNODE * RH) return;
  int n = idx >> 8, j = idx & 255;
  hcat[(size_t)n * 512 + j] = h[idx];
}

// ---------------- GCN scatter-add + readout ----------------

__global__ void scatter_add(const float* __restrict__ h, const int* __restrict__ src,
                            const int* __restrict__ dst, float* __restrict__ agg, int C) {
  int idx = blockIdx.x * blockDim.x + threadIdx.x;
  int cq = C >> 2;
  int e = idx / cq;
  if (e >= NEDGE) return;
  int c = (idx - e * cq) << 2;
  const float* hp = h + (size_t)src[e] * C + c;
  float* ap = agg + (size_t)dst[e] * C + c;
  atomicAdd(ap + 0, hp[0]);
  atomicAdd(ap + 1, hp[1]);
  atomicAdd(ap + 2, hp[2]);
  atomicAdd(ap + 3, hp[3]);
}

__global__ void readout(const float* __restrict__ h, const int* __restrict__ ntype,
                        float* __restrict__ out) {
  int b = blockIdx.x;        // graph
  int c = threadIdx.x;       // feature (256)
  float hist = 0.0f, last = 0.0f, resp = 0.0f, cnt = 0.0f;
  for (int n = 0; n < NPG; ++n) {
    int node = b * NPG + n;
    float v = h[(size_t)node * GHID + c];
    int t = ntype[node];
    if (t == 0)      { hist += v; cnt += 1.0f; }
    else if (t == 1) { last += v; }
    else             { resp += v; }
  }
  out[(size_t)b * GHID + c]            = hist / cnt;
  out[(size_t)(NB + b) * GHID + c]     = last;
  out[(size_t)(2 * NB + b) * GHID + c] = resp;
}

// ---------------- host launch ----------------

extern "C" void kernel_launch(void* const* d_in, const int* in_sizes, int n_in,
                              void* d_out, int out_size, void* d_ws, size_t ws_size,
                              hipStream_t stream) {
  const int*   tokens = (const int*)d_in[0];
  const int*   esrc   = (const int*)d_in[1];
  const int*   edst   = (const int*)d_in[2];
  const int*   ntype  = (const int*)d_in[4];
  const float* emb    = (const float*)d_in[5];
  const float* Wihf   = (const float*)d_in[6];
  const float* Whhf   = (const float*)d_in[7];
  const float* bihf   = (const float*)d_in[8];
  const float* bhhf   = (const float*)d_in[9];
  const float* Wihb   = (const float*)d_in[10];
  const float* Whhb   = (const float*)d_in[11];
  const float* bihb   = (const float*)d_in[12];
  const float* bhhb   = (const float*)d_in[13];
  const float* Wg[4]  = {(const float*)d_in[14], (const float*)d_in[16],
                         (const float*)d_in[18], (const float*)d_in[20]};
  const float* bg[4]  = {(const float*)d_in[15], (const float*)d_in[17],
                         (const float*)d_in[19], (const float*)d_in[21]};
  (void)in_sizes; (void)n_in; (void)out_size; (void)ws_size;

  char* ws = (char*)d_ws;
  size_t off = 0;
  auto alloc = [&](size_t bytes) -> void* {
    void* p = ws + off;
    off = (off + bytes + 255) & ~(size_t)255;
    return p;
  };

  bf16_t* emb_b  = (bf16_t*)alloc((size_t)V_VOCAB * D_PAD * 2);
  bf16_t* Wihf_b = (bf16_t*)alloc((size_t)H3 * D_PAD * 2);
  bf16_t* Wihb_b = (bf16_t*)alloc((size_t)H3 * D_PAD * 2);
  bf16_t* Whhf_b = (bf16_t*)alloc((size_t)H3 * RH * 2);
  bf16_t* Whhb_b = (bf16_t*)alloc((size_t)H3 * RH * 2);
  bf16_t* Wgt[4];
  Wgt[0] = (bf16_t*)alloc((size_t)GHID * 512 * 2);
  Wgt[1] = (bf16_t*)alloc((size_t)GHID * 256 * 2);
  Wgt[2] = (bf16_t*)alloc((size_t)GHID * 256 * 2);
  Wgt[3] = (bf16_t*)alloc((size_t)GHID * 256 * 2);
  float*  xp    = (float*)alloc((size_t)NNODE * H3 * 4);
  float*  gh    = (float*)alloc((size_t)NNODE * H3 * 4);
  float*  hf    = (float*)alloc((size_t)NNODE * RH * 4);
  bf16_t* hb16  = (bf16_t*)alloc((size_t)NNODE * RH * 2);
  float*  hcat  = (float*)alloc((size_t)NNODE * 512 * 4);
  float*  aggf  = (float*)alloc((size_t)NNODE * 512 * 4);
  bf16_t* aggb  = (bf16_t*)alloc((size_t)NNODE * 512 * 2);
  float*  hA    = (float*)alloc((size_t)NNODE * GHID * 4);
  float*  hB    = (float*)alloc((size_t)NNODE * GHID * 4);

  auto nb = [](int n) { return (n + 255) / 256; };

  // ---- weight / table packing (bf16, K-padded, B-operand layouts) ----
  pack_rows_bf16<<<nb(V_VOCAB * D_PAD), 256, 0, stream>>>(emb, emb_b, V_VOCAB, D_IN, D_PAD);
  pack_rows_bf16<<<nb(H3 * D_PAD), 256, 0, stream>>>(Wihf, Wihf_b, H3, D_IN, D_PAD);
  pack_rows_bf16<<<nb(H3 * D_PAD), 256, 0, stream>>>(Wihb, Wihb_b, H3, D_IN, D_PAD);
  pack_rows_bf16<<<nb(H3 * RH), 256, 0, stream>>>(Whhf, Whhf_b, H3, RH, RH);
  pack_rows_bf16<<<nb(H3 * RH), 256, 0, stream>>>(Whhb, Whhb_b, H3, RH, RH);
  pack_transpose_bf16<<<nb(512 * GHID), 256, 0, stream>>>(Wg[0], Wgt[0], 512, GHID);
  for (int l = 1; l < 4; ++l)
    pack_transpose_bf16<<<nb(256 * GHID), 256, 0, stream>>>(Wg[l], Wgt[l], 256, GHID);

  // ---- h0 = 0 ----
  fill_f32<<<nb(NNODE * RH), 256, 0, stream>>>(hf, NNODE * RH);
  fill_bf16<<<nb(NNODE * RH), 256, 0, stream>>>(hb16, NNODE * RH);

  auto gemm = [&](const bf16_t* A, int lda, const int* gat, int gs, int go,
                  const bf16_t* B, int ldb, const float* bias, float* C, int ldc,
                  int M, int Nc, int K, int relu) {
    int tiles = (M / (16 * MT)) * (Nc / (16 * NT));
    wmma_gemm_bf16<<<(tiles + 7) / 8, 256, 0, stream>>>(
        A, lda, gat, gs, go, B, ldb, bias, C, ldc, M, Nc, K, relu);
  };

  // ---- forward GRU scan (fused gather + input projection each step) ----
  for (int t = 0; t < TSEQ; ++t) {
    gemm(emb_b, D_PAD, tokens, TSEQ, t, Wihf_b, D_PAD, bihf, xp, H3,
         NNODE, H3, D_PAD, 0);
    gemm(hb16, RH, nullptr, 0, 0, Whhf_b, RH, bhhf, gh, H3,
         NNODE, H3, RH, 0);
    gru_gate_step<<<nb(NNODE * RH), 256, 0, stream>>>(xp, gh, hf, hb16);
  }
  copy_fwd_to_cat<<<nb(NNODE * RH), 256, 0, stream>>>(hf, hcat);

  // ---- backward GRU: one cell step from zero state on timestep T-1 ----
  gemm(emb_b, D_PAD, tokens, TSEQ, TSEQ - 1, Wihb_b, D_PAD, bihb, xp, H3,
       NNODE, H3, D_PAD, 0);
  gru_gate_bwd<<<nb(NNODE * RH), 256, 0, stream>>>(xp, bhhb, hcat);

  // ---- 4 GCN layers: scatter-add aggregate, then bf16 WMMA GEMM + ReLU ----
  const float* hin = hcat;
  int C = 512;
  float* outs[4] = {hA, hB, hA, hB};
  for (int l = 0; l < 4; ++l) {
    fill_f32<<<nb(NNODE * C), 256, 0, stream>>>(aggf, NNODE * C);
    scatter_add<<<nb(NEDGE * (C / 4)), 256, 0, stream>>>(hin, esrc, edst, aggf, C);
    cast_f32_bf16<<<nb(NNODE * C), 256, 0, stream>>>(aggf, aggb, NNODE * C);
    gemm(aggb, C, nullptr, 0, 0, Wgt[l], C, bg[l], outs[l], GHID,
         NNODE, GHID, C, 1);
    hin = outs[l];
    C = GHID;
  }

  // ---- per-graph readout (hist mean, last, resp) ----
  readout<<<NB, 256, 0, stream>>>(hin, ntype, (float*)d_out);
}
